// DenoisingNet_MLP_29764123361892
// MI455X (gfx1250) — matrix operand order repaired and checked by reading, not compiled, not run on hardware
//
#include <hip/hip_runtime.h>
#include <hip/hip_bf16.h>

typedef __bf16 bf16_t;
typedef __attribute__((ext_vector_type(16))) __bf16 v16bf;
typedef __attribute__((ext_vector_type(8)))  __bf16 v8bf;
typedef __attribute__((ext_vector_type(8)))  float  v8f;

#define M_TOTAL 117128   // 8 * 121 * 121 patch rows
#define LPATCH  14641    // 121 * 121
#define TM      128      // rows per workgroup (16 per wave, 8 waves)

// ---------------- WMMA fragment helpers (wave32, bf16 16x16x32) -------------
// A (16x32, 16-bit) per ISA 7.12.2: lane<16 -> row=lane, K={k0..k0+7, k0+16..k0+23}
//                                   lane>=16 -> row=lane-16, K={k0+8..k0+15, k0+24..k0+31}
__device__ __forceinline__ v16bf ld_frag_a(const bf16_t* A, int lda, int k0, int lane) {
  const int h = (lane >> 4) & 1;
  const int m = lane & 15;
  const bf16_t* p = A + m * lda + k0 + 8 * h;
  v8bf lo = *(const v8bf*)(p);
  v8bf hi = *(const v8bf*)(p + 16);
  return __builtin_shufflevector(lo, hi, 0,1,2,3,4,5,6,7,8,9,10,11,12,13,14,15);
}

// B (32x16, 16-bit): lane<16 -> col=n0+lane, K={k0..k0+7, k0+16..k0+23} ; lane>=16 -> col=n0+lane-16, K={k0+8..k0+15, k0+24..k0+31}
// Loaded from BT where BT[n][k] = B[k][n] (contiguous per lane). For S, BT == S (symmetric).
__device__ __forceinline__ v16bf ld_frag_bt(const bf16_t* BT, int ldbt, int n0, int k0, int lane) {
  const int n  = n0 + (lane & 15);
  const int kk = k0 + 8 * ((lane >> 4) & 1);
  const bf16_t* p = BT + n * ldbt + kk;
  v8bf lo = *(const v8bf*)(p);
  v8bf hi = *(const v8bf*)(p + 16);
  return __builtin_shufflevector(lo, hi, 0,1,2,3,4,5,6,7,8,9,10,11,12,13,14,15);
}

__device__ __forceinline__ v8f wmma_bf16(v16bf a, v16bf b, v8f c) {
  return __builtin_amdgcn_wmma_f32_16x16x32_bf16(false, a, false, b, (short)0, c, false, false);
}

__device__ __forceinline__ float softshrink(float v, float t) {
  float s = fabsf(v) - t;
  return s > 0.0f ? copysignf(s, v) : 0.0f;
}

// ---------------- prep: S = I - (D^T D)/c  (symmetric), bf16 -----------------
__global__ void prep_S(const float* __restrict__ Dict, const float* __restrict__ c,
                       bf16_t* __restrict__ Sg) {
  int a = blockIdx.x;      // 0..255
  int b = threadIdx.x;     // 0..255
  float s = 0.0f;
  #pragma unroll 8
  for (int k = 0; k < 64; ++k)
    s += Dict[k * 256 + a] * Dict[k * 256 + b];
  float v = (a == b ? 1.0f : 0.0f) - s / c[0];
  Sg[a * 256 + b] = (bf16_t)v;
}

// ---------------- prep: bf16 transposed weight copies ------------------------
__global__ void prep_W(const float* __restrict__ Dict, const float* __restrict__ W1,
                       const float* __restrict__ W2, const float* __restrict__ W3,
                       bf16_t* __restrict__ DictTg, bf16_t* __restrict__ Dictbf,
                       bf16_t* __restrict__ W1Tg, bf16_t* __restrict__ W2Tg,
                       bf16_t* __restrict__ W3Tg) {
  int idx = blockIdx.x * 256 + threadIdx.x;   // 0..16383
  { // DictT[256][64], Dictbf[64][256]
    int l = idx >> 6, q = idx & 63;
    DictTg[idx] = (bf16_t)Dict[q * 256 + l];
    Dictbf[idx] = (bf16_t)Dict[idx];
  }
  if (idx < 8192) { int n = idx >> 6, k = idx & 63;  W1Tg[idx] = (bf16_t)W1[k * 128 + n]; } // W1T[128][64]
  if (idx < 8192) { int n = idx >> 7, k = idx & 127; W2Tg[idx] = (bf16_t)W2[k * 64  + n]; } // W2T[64][128]
  if (idx < 2048) { int n = idx >> 6, k = idx & 63;  W3Tg[idx] = (bf16_t)W3[k * 32  + n]; } // W3T[32][64]
}

__global__ void zero_out(float* __restrict__ out) {
  int i = blockIdx.x * 256 + threadIdx.x;
  if (i < 131072) out[i] = 0.0f;
}

__global__ void norm_out(float* __restrict__ out, const float* __restrict__ w) {
  int idx = blockIdx.x * 256 + threadIdx.x;
  if (idx < 131072) {
    int i = (idx >> 7) & 127, j = idx & 127;
    int ci = (i < 7 ? i : 7) - (i - 120 > 0 ? i - 120 : 0) + 1;  // patches covering row i
    int cj = (j < 7 ? j : 7) - (j - 120 > 0 ? j - 120 : 0) + 1;
    out[idx] = out[idx] / (w[0] * (float)(ci * cj));
  }
}

// ---------------- fused main kernel ------------------------------------------
// Dynamic LDS layout (bytes), total 262656 (<= 320KB/WGP):
//   [0      , 131072) transient: phase1 {U 16K | h1 32K | h2 16K | h3f 16K} then S(bf16) 128K
//   [131072 , 196608) yT bf16 256x128 (transposed: yT[col][row])
//   [196608 , 262144) z  bf16 128x256 (row-major, A-operand source)
//   [262144 , 262656) thr f32 x128
__global__ __launch_bounds__(256) void lista_main(
    const float* __restrict__ x,  const float* __restrict__ c,  const float* __restrict__ w,
    const float* __restrict__ b1, const float* __restrict__ b2, const float* __restrict__ b3,
    const float* __restrict__ W4, const float* __restrict__ b4,
    const bf16_t* __restrict__ Sg,    const bf16_t* __restrict__ DictTg,
    const bf16_t* __restrict__ Dictbf,const bf16_t* __restrict__ W1Tg,
    const bf16_t* __restrict__ W2Tg,  const bf16_t* __restrict__ W3Tg,
    float* __restrict__ out)
{
  extern __shared__ __align__(16) char smem[];
  bf16_t* sU   = (bf16_t*)(smem);            // 128x64
  bf16_t* sH1  = (bf16_t*)(smem + 16384);    // 128x128
  bf16_t* sH2  = (bf16_t*)(smem + 49152);    // 128x64
  float*  sH3  = (float*) (smem + 65536);    // 128x32 f32
  bf16_t* sS   = (bf16_t*)(smem);            // 256x256 (phase 2, overlays phase-1 region)
  bf16_t* sYT  = (bf16_t*)(smem + 131072);   // 256x128 (col-major y)
  bf16_t* sZ   = (bf16_t*)(smem + 196608);   // 128x256
  float*  sThr = (float*) (smem + 262144);   // 128

  const int tid   = threadIdx.x;
  const int lane  = tid & 31;
  const int wave  = tid >> 5;
  const int wrow0 = wave * 16;               // this wave owns rows [wrow0, wrow0+16)
  const int tile0 = blockIdx.x * TM;
  const float invc = 1.0f / c[0];
  const float wv   = w[0];
  const int colL = lane & 15;
  const int rb   = 8 * ((lane >> 4) & 1);

  // Warm S toward L2 early (global_prefetch_b8); consumed after phase 1.
  __builtin_prefetch(Sg + tid * 256, 0, 1);
  __builtin_prefetch(Sg + tid * 256 + 128, 0, 1);

  // ---- unfold: load this tile's 128 patch rows (64 elems each) into LDS bf16
  {
    int r = tid >> 1;
    int g = tile0 + r;
    bool valid = g < M_TOTAL;
    int gg  = valid ? g : 0;
    int n   = gg / LPATCH;
    int rem = gg - n * LPATCH;
    int pi  = rem / 121;
    int pj  = rem - pi * 121;
    const float* bx = x + n * 16384 + pi * 128 + pj;
    int d0 = (tid & 1) * 4;                 // this thread covers di = d0..d0+3
    #pragma unroll
    for (int dd = 0; dd < 4; ++dd) {
      int di = d0 + dd;
      v8bf pk;
      #pragma unroll
      for (int j = 0; j < 8; ++j) {
        float v = valid ? bx[di * 128 + j] : 0.0f;
        pk[j] = (bf16_t)v;
      }
      *(v8bf*)(sU + r * 64 + di * 8) = pk;   // one ds_store_b128
    }
  }
  __syncthreads();

  const bf16_t* Au = sU + wrow0 * 64;

  // ---- MLP: h1 = relu(U@W1 + b1)   [16x64]@[64x128]
  {
    v8f acc[8];
    const v8f z8 = {0.f,0.f,0.f,0.f,0.f,0.f,0.f,0.f};
    #pragma unroll
    for (int nt = 0; nt < 8; ++nt) acc[nt] = z8;
    #pragma unroll
    for (int kb = 0; kb < 2; ++kb) {
      v16bf a = ld_frag_a(Au, 64, kb * 32, lane);
      #pragma unroll
      for (int nt = 0; nt < 8; ++nt) {
        v16bf b = ld_frag_bt(W1Tg, 64, nt * 16, kb * 32, lane);
        acc[nt] = wmma_bf16(a, b, acc[nt]);
      }
    }
    #pragma unroll
    for (int nt = 0; nt < 8; ++nt) {
      int col = nt * 16 + colL;
      float bias = b1[col];
      #pragma unroll
      for (int r = 0; r < 8; ++r) {
        float v = acc[nt][r] + bias; v = v > 0.f ? v : 0.f;
        sH1[(wrow0 + rb + r) * 128 + col] = (bf16_t)v;
      }
    }
  }

  // ---- h2 = relu(h1@W2 + b2)   [16x128]@[128x64]
  const bf16_t* Ah1 = sH1 + wrow0 * 128;
  {
    v8f acc[4];
    const v8f z8 = {0.f,0.f,0.f,0.f,0.f,0.f,0.f,0.f};
    #pragma unroll
    for (int nt = 0; nt < 4; ++nt) acc[nt] = z8;
    #pragma unroll
    for (int kb = 0; kb < 4; ++kb) {
      v16bf a = ld_frag_a(Ah1, 128, kb * 32, lane);
      #pragma unroll
      for (int nt = 0; nt < 4; ++nt) {
        v16bf b = ld_frag_bt(W2Tg, 128, nt * 16, kb * 32, lane);
        acc[nt] = wmma_bf16(a, b, acc[nt]);
      }
    }
    #pragma unroll
    for (int nt = 0; nt < 4; ++nt) {
      int col = nt * 16 + colL;
      float bias = b2[col];
      #pragma unroll
      for (int r = 0; r < 8; ++r) {
        float v = acc[nt][r] + bias; v = v > 0.f ? v : 0.f;
        sH2[(wrow0 + rb + r) * 64 + col] = (bf16_t)v;
      }
    }
  }

  // ---- h3 = relu(h2@W3 + b3)   [16x64]@[64x32]  -> f32 LDS
  const bf16_t* Ah2 = sH2 + wrow0 * 64;
  {
    v8f acc[2];
    const v8f z8 = {0.f,0.f,0.f,0.f,0.f,0.f,0.f,0.f};
    acc[0] = z8; acc[1] = z8;
    #pragma unroll
    for (int kb = 0; kb < 2; ++kb) {
      v16bf a = ld_frag_a(Ah2, 64, kb * 32, lane);
      #pragma unroll
      for (int nt = 0; nt < 2; ++nt) {
        v16bf b = ld_frag_bt(W3Tg, 64, nt * 16, kb * 32, lane);
        acc[nt] = wmma_bf16(a, b, acc[nt]);
      }
    }
    #pragma unroll
    for (int nt = 0; nt < 2; ++nt) {
      int col = nt * 16 + colL;
      float bias = b3[col];
      #pragma unroll
      for (int r = 0; r < 8; ++r) {
        float v = acc[nt][r] + bias; v = v > 0.f ? v : 0.f;
        sH3[(wrow0 + rb + r) * 32 + col] = v;
      }
    }
  }

  // ---- lambda = h3@W4 + b4 ; thr = lambda / c   (per-wave, lanes 0..15)
  if (lane < 16) {
    int row = wrow0 + lane;
    float s = b4[0];
    #pragma unroll
    for (int k = 0; k < 32; ++k) s += sH3[row * 32 + k] * W4[k];
    sThr[row] = s * invc;
  }

  // ---- y = U@Dict ; z0 = soft(y, thr)   [16x64]@[64x256]
  {
    v8f acc[16];
    const v8f z8 = {0.f,0.f,0.f,0.f,0.f,0.f,0.f,0.f};
    #pragma unroll
    for (int nt = 0; nt < 16; ++nt) acc[nt] = z8;
    #pragma unroll
    for (int kb = 0; kb < 2; ++kb) {
      v16bf a = ld_frag_a(Au, 64, kb * 32, lane);
      #pragma unroll
      for (int nt = 0; nt < 16; ++nt) {
        v16bf b = ld_frag_bt(DictTg, 64, nt * 16, kb * 32, lane);
        acc[nt] = wmma_bf16(a, b, acc[nt]);
      }
    }
    #pragma unroll
    for (int nt = 0; nt < 16; ++nt) {
      int col = nt * 16 + colL;
      v8bf ypk;
      #pragma unroll
      for (int r = 0; r < 8; ++r) {
        int row = wrow0 + rb + r;
        float yv = acc[nt][r];
        ypk[r] = (bf16_t)yv;
        sZ[row * 256 + col] = (bf16_t)softshrink(yv, sThr[row]);
      }
      // transposed y store: rows contiguous -> single ds_store_b128
      *(v8bf*)(sYT + col * 128 + wrow0 + rb) = ypk;
    }
  }

  __syncthreads();  // phase-1 transient region free; all waves done

  // ---- cooperative stage of S (256x256 bf16, 128KB) into LDS
  {
    const v8bf* src = (const v8bf*)Sg;
    v8bf*       dst = (v8bf*)sS;
    #pragma unroll 4
    for (int i = 0; i < 32; ++i) dst[tid + i * 256] = src[tid + i * 256];
  }
  __syncthreads();

  // hoist per-lane thresholds (iteration-invariant): rows wrow0+rb+0..7
  float thrv[8];
  #pragma unroll
  for (int r = 0; r < 8; ++r) thrv[r] = sThr[wrow0 + rb + r];

  // ---- LISTA: 7x { z = soft(z@S + y/c, thr) }  (row-independent per wave)
  const bf16_t* Az = sZ + wrow0 * 256;
  for (int it = 0; it < 7; ++it) {
    v8f acc[16];
    const v8f z8 = {0.f,0.f,0.f,0.f,0.f,0.f,0.f,0.f};
    #pragma unroll
    for (int nt = 0; nt < 16; ++nt) acc[nt] = z8;
    for (int kb = 0; kb < 8; ++kb) {
      v16bf a = ld_frag_a(Az, 256, kb * 32, lane);
      #pragma unroll
      for (int nt = 0; nt < 16; ++nt) {
        v16bf b = ld_frag_bt(sS, 256, nt * 16, kb * 32, lane);
        acc[nt] = wmma_bf16(a, b, acc[nt]);
      }
    }
    #pragma unroll
    for (int nt = 0; nt < 16; ++nt) {
      int col = nt * 16 + colL;
      v8bf ypk = *(const v8bf*)(sYT + col * 128 + wrow0 + rb);  // one ds_load_b128
      #pragma unroll
      for (int r = 0; r < 8; ++r) {
        int row = wrow0 + rb + r;
        float v = acc[nt][r] + (float)ypk[r] * invc;
        sZ[row * 256 + col] = (bf16_t)softshrink(v, thrv[r]);
      }
    }
  }

  // ---- decode: xpred = clip(z@Dict^T, 0, 1)*w ; fold via atomics
  // B = Dict^T, so BT[n][k] = Dict[n][k] -> row-major Dict bf16 directly.
  {
    v8f acc[4];
    const v8f z8 = {0.f,0.f,0.f,0.f,0.f,0.f,0.f,0.f};
    #pragma unroll
    for (int nt = 0; nt < 4; ++nt) acc[nt] = z8;
    #pragma unroll
    for (int kb = 0; kb < 8; ++kb) {
      v16bf a = ld_frag_a(Az, 256, kb * 32, lane);
      #pragma unroll
      for (int nt = 0; nt < 4; ++nt) {
        v16bf b = ld_frag_bt(Dictbf, 256, nt * 16, kb * 32, lane);
        acc[nt] = wmma_bf16(a, b, acc[nt]);
      }
    }
    #pragma unroll
    for (int nt = 0; nt < 4; ++nt) {
      int e  = nt * 16 + colL;
      int di = e >> 3, dj = e & 7;
      #pragma unroll
      for (int r = 0; r < 8; ++r) {
        int g = tile0 + wrow0 + rb + r;
        if (g < M_TOTAL) {
          float v = acc[nt][r];
          v = v < 0.f ? 0.f : (v > 1.f ? 1.f : v);
          v *= wv;
          int n   = g / LPATCH;
          int rem = g - n * LPATCH;
          int pi  = rem / 121;
          int pj  = rem - pi * 121;
          atomicAdd(out + n * 16384 + (pi + di) * 128 + (pj + dj), v);
        }
      }
    }
  }
}

// ---------------- host launcher ----------------------------------------------
extern "C" void kernel_launch(void* const* d_in, const int* in_sizes, int n_in,
                              void* d_out, int out_size, void* d_ws, size_t ws_size,
                              hipStream_t stream) {
  const float* x    = (const float*)d_in[0];
  const float* Dict = (const float*)d_in[1];
  const float* c    = (const float*)d_in[2];
  const float* w    = (const float*)d_in[3];
  const float* W1   = (const float*)d_in[4];
  const float* b1   = (const float*)d_in[5];
  const float* W2   = (const float*)d_in[6];
  const float* b2   = (const float*)d_in[7];
  const float* W3   = (const float*)d_in[8];
  const float* b3   = (const float*)d_in[9];
  const float* W4   = (const float*)d_in[10];
  const float* b4   = (const float*)d_in[11];
  float* out = (float*)d_out;

  char* ws = (char*)d_ws;   // needs ~228KB: bf16 S / DictT / Dict / W1T / W2T / W3T
  bf16_t* Sg     = (bf16_t*)(ws);
  bf16_t* DictTg = (bf16_t*)(ws + 131072);
  bf16_t* Dictbf = (bf16_t*)(ws + 163840);
  bf16_t* W1Tg   = (bf16_t*)(ws + 196608);
  bf16_t* W2Tg   = (bf16_t*)(ws + 212992);
  bf16_t* W3Tg   = (bf16_t*)(ws + 229376);

  zero_out<<<512, 256, 0, stream>>>(out);
  prep_S<<<256, 256, 0, stream>>>(Dict, c, Sg);
  prep_W<<<64, 256, 0, stream>>>(Dict, W1, W2, W3, DictTg, Dictbf, W1Tg, W2Tg, W3Tg);

  const int tiles = (M_TOTAL + TM - 1) / TM;   // 916
  lista_main<<<tiles, 256, 262656, stream>>>(x, c, w, b1, b2, b3, W4, b4,
                                             Sg, DictTg, Dictbf, W1Tg, W2Tg, W3Tg, out);
  norm_out<<<512, 256, 0, stream>>>(out, w);
}
